// ODEfunc_16690242912383
// MI455X (gfx1250) — matrix operand order repaired and checked by reading, not compiled, not run on hardware
//
#include <hip/hip_runtime.h>

// ---------- types ----------
typedef _Float16 h8   __attribute__((ext_vector_type(8)));
typedef _Float16 v16h __attribute__((ext_vector_type(16)));
typedef float    v8f  __attribute__((ext_vector_type(8)));
typedef float    f4   __attribute__((ext_vector_type(4)));

// ---------- LDS layout (bytes) ----------
// weights (f16), shared by whole block:
//   W0f : W0[:,1:]  (256 x  64)  @ 0       (32768 B)
//   W1f : W1[:,1:]  (256 x 256)  @ 32768   (131072 B)
//   W2f : W2[:,1:]  ( 64 x 256)  @ 163840  (32768 B)
// effective biases (f32): b + t*W[:,0]
//   b0e[256] @ 196608, b1e[256] @ 197632, b2e[64] @ 198656
// per-PAIR region @ 198912 + pair*28672 (two waves cooperate on 16 rows):
//   a0T (256x16 f16) +0, bufB (a1T then uT, 256x16) +8192,
//   g1  (16x256 f16 row-major) +16384, yT (64x16, later f32 div scratch) +24576,
//   eT (64x16) +26624
#define LDS_TOTAL 313600

__device__ __forceinline__ v16h cat16(h8 lo, h8 hi) {
  return __builtin_shufflevector(lo, hi, 0,1,2,3,4,5,6,7,8,9,10,11,12,13,14,15);
}

// A-fragment (16xK tile, row-major f16 in LDS): lane holds row = lane&15,
// k-offsets (lane>>4)*8 + {0..7} and +16 (per CDNA5 16-bit A layout).
__device__ __forceinline__ v16h loadA(const _Float16* base, int ld, int r0, int k0, int lane) {
  const _Float16* p = base + (r0 + (lane & 15)) * ld + k0 + ((lane >> 4) << 3);
  return cat16(*(const h8*)p, *(const h8*)(p + 16));
}

// B-fragment (Kx16 tile, row-major f16 in LDS): lane holds n = n0 + (lane&15),
// k-offsets (lane>>4)*8 + {0..7} and +16 (matching the A layout).
__device__ __forceinline__ v16h loadB(const _Float16* base, int ld, int k0, int n0, int lane) {
  const _Float16* p = base + (k0 + ((lane >> 4) << 3)) * ld + n0 + (lane & 15);
  v16h r;
#pragma unroll
  for (int s = 0; s < 8; ++s) {
    r[s]     = p[s * ld];
    r[8 + s] = p[(16 + s) * ld];
  }
  return r;
}

// A-fragment straight from row-major fp32 global memory (converted to f16).
__device__ __forceinline__ v16h loadA_g32(const float* base, int ld, int r0, int k0, int lane) {
  const float* p = base + (r0 + (lane & 15)) * ld + k0 + ((lane >> 4) << 3);
  f4 x0 = *(const f4*)p;        f4 x1 = *(const f4*)(p + 4);
  f4 x2 = *(const f4*)(p + 16); f4 x3 = *(const f4*)(p + 20);
  v16h r;
#pragma unroll
  for (int j = 0; j < 4; ++j) {
    r[j]      = (_Float16)x0[j];
    r[4 + j]  = (_Float16)x1[j];
    r[8 + j]  = (_Float16)x2[j];
    r[12 + j] = (_Float16)x3[j];
  }
  return r;
}

__device__ __forceinline__ v8f wmma(v16h a, v16h b, v8f c) {
  return __builtin_amdgcn_wmma_f32_16x16x32_f16(false, a, false, b, (short)0, c, false, false);
}

// Branchless fast softplus on hardware transcendentals:
//   softplus(z) = max(z,0) + log(1 + exp(-|z|))
__device__ __forceinline__ float softplus(float z) {
  return fmaxf(z, 0.f) + __logf(1.f + __expf(-fabsf(z)));
}

__global__ __launch_bounds__(256, 1)
void odefunc_kernel(const float* __restrict__ t,  const float* __restrict__ y,
                    const float* __restrict__ e,
                    const float* __restrict__ W0, const float* __restrict__ b0,
                    const float* __restrict__ W1, const float* __restrict__ b1,
                    const float* __restrict__ W2, const float* __restrict__ b2,
                    float* __restrict__ out) {
  extern __shared__ char smem[];
  _Float16* w0f = (_Float16*)(smem + 0);
  _Float16* w1f = (_Float16*)(smem + 32768);
  _Float16* w2f = (_Float16*)(smem + 163840);
  float*    b0e = (float*)(smem + 196608);
  float*    b1e = (float*)(smem + 197632);
  float*    b2e = (float*)(smem + 198656);

  const int tid = threadIdx.x;
  const float t0 = t[0];

  // ---- stage weights (f16) + effective biases into LDS ----
  for (int i = tid; i < 256 * 64;  i += 256) w0f[i] = (_Float16)W0[(i >> 6) * 65  + 1 + (i & 63)];
  for (int i = tid; i < 256 * 256; i += 256) w1f[i] = (_Float16)W1[(i >> 8) * 257 + 1 + (i & 255)];
  for (int i = tid; i < 64 * 256;  i += 256) w2f[i] = (_Float16)W2[(i >> 8) * 257 + 1 + (i & 255)];
  if (tid < 256) {
    b0e[tid] = b0[tid] + t0 * W0[tid * 65];
    b1e[tid] = b1[tid] + t0 * W1[tid * 257];
  }
  if (tid < 64) b2e[tid] = b2[tid] + t0 * W2[tid * 257];
  __syncthreads();

  const int wave = tid >> 5, lane = tid & 31;
  const int pair = wave >> 1, sub = wave & 1;      // two waves per 16-row tile
  const int l16 = lane & 15, lh = lane >> 4;
  const int row0 = (blockIdx.x * 4 + pair) * 16;

  char* pw = smem + 198912 + pair * 28672;
  _Float16* a0T  = (_Float16*)(pw + 0);      // 256 x 16 : a0 transposed [k][m]
  _Float16* bufB = (_Float16*)(pw + 8192);   // a1T, later uT
  _Float16* g1m  = (_Float16*)(pw + 16384);  // 16 x 256 row-major [m][k]
  _Float16* yT   = (_Float16*)(pw + 24576);  // 64 x 16 (later: f32 div scratch)
  _Float16* eT   = (_Float16*)(pw + 26624);  // 64 x 16

  // ---- load input tiles transposed (f16); pair of waves splits the copy ----
  for (int i = lane + sub * 32; i < 1024; i += 64) {
    int m = i >> 6, k = i & 63;
    yT[k * 16 + m] = (_Float16)y[(row0 + m) * 65 + k];
    eT[k * 16 + m] = (_Float16)e[(row0 + m) * 64 + k];
  }
  __syncthreads();

  v8f c[8];

  // ---- L0 forward (transposed form): z0^T = W0f_rows x yT ; 8 n-tiles/wave ----
#pragma unroll
  for (int j = 0; j < 8; ++j)
#pragma unroll
    for (int i = 0; i < 8; ++i) c[j][i] = b0e[(sub * 8 + j) * 16 + i + 8 * lh];
#pragma unroll
  for (int ks = 0; ks < 2; ++ks) {
    v16h b = loadB(yT, 16, ks * 32, 0, lane);          // invariant over n-tiles
#pragma unroll
    for (int j = 0; j < 8; ++j)
      c[j] = wmma(loadA(w0f, 64, (sub * 8 + j) * 16, ks * 32, lane), b, c[j]);
  }
#pragma unroll
  for (int j = 0; j < 8; ++j)
#pragma unroll
    for (int i = 0; i < 8; ++i)
      a0T[((sub * 8 + j) * 16 + i + 8 * lh) * 16 + l16] = (_Float16)softplus(c[j][i]);
  __syncthreads();

  // ---- L1 forward: z1^T = W1f_rows x a0T ----
#pragma unroll
  for (int j = 0; j < 8; ++j)
#pragma unroll
    for (int i = 0; i < 8; ++i) c[j][i] = b1e[(sub * 8 + j) * 16 + i + 8 * lh];
#pragma unroll
  for (int ks = 0; ks < 8; ++ks) {
    v16h b = loadB(a0T, 16, ks * 32, 0, lane);
#pragma unroll
    for (int j = 0; j < 8; ++j)
      c[j] = wmma(loadA(w1f, 256, (sub * 8 + j) * 16, ks * 32, lane), b, c[j]);
  }
#pragma unroll
  for (int j = 0; j < 8; ++j)
#pragma unroll
    for (int i = 0; i < 8; ++i)
      bufB[((sub * 8 + j) * 16 + i + 8 * lh) * 16 + l16] = (_Float16)softplus(c[j][i]); // a1T
  __syncthreads();

  // ---- L2 forward: dx^T = W2f_rows x a1T ; 2 n-tiles/wave; write dx to out ----
#pragma unroll
  for (int j = 0; j < 2; ++j)
#pragma unroll
    for (int i = 0; i < 8; ++i) c[j][i] = b2e[(sub * 2 + j) * 16 + i + 8 * lh];
#pragma unroll
  for (int ks = 0; ks < 8; ++ks) {
    v16h b = loadB(bufB, 16, ks * 32, 0, lane);
#pragma unroll
    for (int j = 0; j < 2; ++j)
      c[j] = wmma(loadA(w2f, 256, (sub * 2 + j) * 16, ks * 32, lane), b, c[j]);
  }
#pragma unroll
  for (int j = 0; j < 2; ++j)
#pragma unroll
    for (int i = 0; i < 8; ++i)
      out[(row0 + l16) * 65 + (sub * 2 + j) * 16 + i + 8 * lh] = c[j][i];

  // ---- backward L2: g1 = (e @ W2f) * sigma(z1), sigma = 1 - exp(-a1) ----
#pragma unroll
  for (int j = 0; j < 8; ++j)
#pragma unroll
    for (int i = 0; i < 8; ++i) c[j][i] = 0.f;
#pragma unroll
  for (int ks = 0; ks < 2; ++ks) {
    v16h a = loadA_g32(e, 64, row0, ks * 32, lane);    // invariant over n-tiles
#pragma unroll
    for (int j = 0; j < 8; ++j)
      c[j] = wmma(a, loadB(w2f, 256, ks * 32, (sub * 8 + j) * 16, lane), c[j]);
  }
#pragma unroll
  for (int j = 0; j < 8; ++j) {
    h8 a1v = *(const h8*)(bufB + ((sub * 8 + j) * 16 + l16) * 16 + 8 * lh);
#pragma unroll
    for (int i = 0; i < 8; ++i) {
      float s1 = 1.f - __expf(-(float)a1v[i]);
      g1m[(i + 8 * lh) * 256 + (sub * 8 + j) * 16 + l16] = (_Float16)(c[j][i] * s1);
    }
  }
  __syncthreads();   // g1 complete; a1T reads done -> bufB reusable

  // ---- u^T = W0f_rows x eT  (u = e @ W0[:,1:]^T); overwrite bufB with uT ----
#pragma unroll
  for (int j = 0; j < 8; ++j)
#pragma unroll
    for (int i = 0; i < 8; ++i) c[j][i] = 0.f;
#pragma unroll
  for (int ks = 0; ks < 2; ++ks) {
    v16h b = loadB(eT, 16, ks * 32, 0, lane);
#pragma unroll
    for (int j = 0; j < 8; ++j)
      c[j] = wmma(loadA(w0f, 64, (sub * 8 + j) * 16, ks * 32, lane), b, c[j]);
  }
#pragma unroll
  for (int j = 0; j < 8; ++j)
#pragma unroll
    for (int i = 0; i < 8; ++i)
      bufB[((sub * 8 + j) * 16 + i + 8 * lh) * 16 + l16] = (_Float16)c[j][i]; // uT[k][m]

  // ---- backward L1 + divergence: g0 = (g1 @ W1f)*sigma(z0); div = sum g0*u ----
#pragma unroll
  for (int j = 0; j < 8; ++j)
#pragma unroll
    for (int i = 0; i < 8; ++i) c[j][i] = 0.f;
#pragma unroll
  for (int ks = 0; ks < 8; ++ks) {
    v16h a = loadA(g1m, 256, 0, ks * 32, lane);        // invariant over n-tiles
#pragma unroll
    for (int j = 0; j < 8; ++j)
      c[j] = wmma(a, loadB(w1f, 256, ks * 32, (sub * 8 + j) * 16, lane), c[j]);
  }
  float dv[8];
#pragma unroll
  for (int i = 0; i < 8; ++i) dv[i] = 0.f;
#pragma unroll
  for (int j = 0; j < 8; ++j) {
    h8 a0v = *(const h8*)(a0T  + ((sub * 8 + j) * 16 + l16) * 16 + 8 * lh);
    h8 uv  = *(const h8*)(bufB + ((sub * 8 + j) * 16 + l16) * 16 + 8 * lh);
#pragma unroll
    for (int i = 0; i < 8; ++i) {
      float s0 = 1.f - __expf(-(float)a0v[i]);
      dv[i] += (c[j][i] * s0) * (float)uv[i];
    }
  }

  // ---- reduce over 16-lane halves, then combine wave pair via LDS scratch ----
  float* dvp = (float*)yT;   // yT is dead; 32 floats of scratch per pair
#pragma unroll
  for (int i = 0; i < 8; ++i) {
    float v = dv[i];
    v += __shfl_xor(v, 1, 32);
    v += __shfl_xor(v, 2, 32);
    v += __shfl_xor(v, 4, 32);
    v += __shfl_xor(v, 8, 32);
    if (l16 == 0) dvp[sub * 16 + i + 8 * lh] = v;
  }
  __syncthreads();
  if (sub == 0 && l16 == 0) {
#pragma unroll
    for (int i = 0; i < 8; ++i) {
      int m = i + 8 * lh;
      out[(row0 + m) * 65 + 64] = -(dvp[m] + dvp[16 + m]);
    }
  }
}

extern "C" void kernel_launch(void* const* d_in, const int* in_sizes, int n_in,
                              void* d_out, int out_size, void* d_ws, size_t ws_size,
                              hipStream_t stream) {
  const float* t  = (const float*)d_in[0];
  const float* y  = (const float*)d_in[1];
  const float* e  = (const float*)d_in[2];
  const float* W0 = (const float*)d_in[3];
  const float* b0 = (const float*)d_in[4];
  const float* W1 = (const float*)d_in[5];
  const float* b1 = (const float*)d_in[6];
  const float* W2 = (const float*)d_in[7];
  const float* b2 = (const float*)d_in[8];
  float* out = (float*)d_out;

  // B = 262144 rows, 64 rows per block (4 pairs x 16 rows, 8 waves) -> 4096 blocks
  odefunc_kernel<<<4096, 256, LDS_TOTAL, stream>>>(t, y, e, W0, b0, W1, b1, W2, b2, out);
}
